// CustomS5Block_80152679678390
// MI455X (gfx1250) — matrix operand order repaired and checked by reading, not compiled, not run on hardware
//
#include <hip/hip_runtime.h>

// ---------------------------------------------------------------------------
// S5 block for MI455X (gfx1250, wave32, WMMA bf16).
//   u = x*w1 + b1  (rank-1)  =>  Bu[b,l,p] = x[b,l]*vw[p] + vb[p]
//   scan: s_l = lam_bar * s_{l-1} + Bu_l   (constant coeff -> chunked scan)
//   s_out = Xcat[32768,512] @ W1[256,512]  (W1 = [2Cre | -2Cim], [N,K] layout)
//   h1 = tanh(s_out + D*u) + u
//   f  = tanh(h1 @ Wenc^T + benc)          (bf16 WMMA, hw v_tanh)
//   h3 = f @ Wdec^T + bdec + h1            (bf16 WMMA, fp32 out)
//   out = h3 @ l2w + l2b
// ---------------------------------------------------------------------------

typedef __attribute__((ext_vector_type(16))) __bf16 bf16x16;
typedef __attribute__((ext_vector_type(8)))  float  floatx8;

#define DIMH  256
#define PDIM  256
#define BATCH 8
#define SEQ   4096
#define ROWS  (BATCH * SEQ)   // 32768
#define HID   (10 * DIMH)     // 2560
#define CHUNK 64
#define NCHUNK (SEQ / CHUNK)  // 64

// Hardware tanh (CDNA5 lists TANH among transcendental VALU ops).
__device__ inline float fast_tanh(float x)
{
#if __has_builtin(__builtin_amdgcn_tanhf)
    return __builtin_amdgcn_tanhf(x);
#elif __has_builtin(__builtin_amdgcn_tanh_f32)
    return __builtin_amdgcn_tanh_f32(x);
#else
    return tanhf(x);
#endif
}

// ---------------------------------------------------------------------------
// Precompute: lam_bar, vw = coef*(B@w1), vb = coef*(B@b1), aT = lam_bar^64
// cst layout (floats): [0]=lbr [256]=lbi [512]=vwr [768]=vwi
//                      [1024]=vbr [1280]=vbi [1536]=aTr [1792]=aTi
// ---------------------------------------------------------------------------
__global__ __launch_bounds__(256) void prep_consts(
    const float* __restrict__ lam_re, const float* __restrict__ lam_im,
    const float* __restrict__ log_step,
    const float* __restrict__ B_re, const float* __restrict__ B_im,
    const float* __restrict__ l1w, const float* __restrict__ l1b,
    float* __restrict__ cst)
{
    int p = threadIdx.x;
    float lr = lam_re[p], li = lam_im[p];
    float step = expf(log_step[p]);
    float ea  = expf(lr * step);
    float lbr = ea * cosf(li * step);
    float lbi = ea * sinf(li * step);
    // coef = (lam_bar - 1) / lam
    float den = lr * lr + li * li;
    float nr = lbr - 1.0f, ni = lbi;
    float cr = (nr * lr + ni * li) / den;
    float ci = (ni * lr - nr * li) / den;
    // sums over h
    float swr = 0.f, swi = 0.f, sbr = 0.f, sbi = 0.f;
    for (int h = 0; h < DIMH; ++h) {
        float br = B_re[p * DIMH + h], bi = B_im[p * DIMH + h];
        float w = l1w[h], b = l1b[h];
        swr = fmaf(br, w, swr); swi = fmaf(bi, w, swi);
        sbr = fmaf(br, b, sbr); sbi = fmaf(bi, b, sbi);
    }
    float vwr = cr * swr - ci * swi, vwi = cr * swi + ci * swr;
    float vbr = cr * sbr - ci * sbi, vbi = cr * sbi + ci * sbr;
    // aT = lam_bar^64 via 6 squarings
    float tr = lbr, ti = lbi;
    for (int q = 0; q < 6; ++q) { float a = tr*tr - ti*ti, b = 2.f*tr*ti; tr = a; ti = b; }
    cst[p]        = lbr;  cst[256 + p]  = lbi;
    cst[512 + p]  = vwr;  cst[768 + p]  = vwi;
    cst[1024 + p] = vbr;  cst[1280 + p] = vbi;
    cst[1536 + p] = tr;   cst[1792 + p] = ti;
}

// W1[h, c] : c<256 -> 2*C_re[h,c] ; c>=256 -> -2*C_im[h,c-256]   (bf16, ld=512)
__global__ __launch_bounds__(256) void build_w1(
    const float* __restrict__ Cre, const float* __restrict__ Cim,
    __bf16* __restrict__ W1)
{
    int i = blockIdx.x * 256 + threadIdx.x;     // 256*512 total
    int h = i >> 9, c = i & 511;
    float v = (c < 256) ? 2.f * Cre[h * 256 + c] : -2.f * Cim[h * 256 + (c - 256)];
    W1[i] = (__bf16)v;
}

__global__ __launch_bounds__(256) void f32_to_bf16(
    const float* __restrict__ in, __bf16* __restrict__ out, int n)
{
    int i = blockIdx.x * 256 + threadIdx.x;
    if (i < n) out[i] = (__bf16)in[i];
}

// ---------------------------------------------------------------------------
// Scan pass 1: per (chunk=blockIdx.x, batch=blockIdx.y, p=tid) compute chunk-
// final state starting from 0. carry[(b*256+p)*64 + chunk] = final (re,im)
// ---------------------------------------------------------------------------
__global__ __launch_bounds__(256) void s5_scan_chunks(
    const float* __restrict__ x, const float* __restrict__ cst,
    float2* __restrict__ carry)
{
    int p = threadIdx.x, chunk = blockIdx.x, b = blockIdx.y;
    float lbr = cst[p], lbi = cst[256 + p];
    float vwr = cst[512 + p], vwi = cst[768 + p];
    float vbr = cst[1024 + p], vbi = cst[1280 + p];
    const float* xp = x + b * SEQ + chunk * CHUNK;
    float sr = 0.f, si = 0.f;
    for (int t = 0; t < CHUNK; ++t) {
        float xv = xp[t];
        float br = fmaf(xv, vwr, vbr);
        float bi = fmaf(xv, vwi, vbi);
        float nr = fmaf(lbr, sr, fmaf(-lbi, si, br));
        float ni = fmaf(lbr, si, fmaf( lbi, sr, bi));
        sr = nr; si = ni;
    }
    carry[(size_t)(b * PDIM + p) * NCHUNK + chunk] = make_float2(sr, si);
}

// Scan pass 2: serial carry composition per chain. prefix[j] = state entering
// chunk j (exclusive scan): c' = aT*c + carry[j]
__global__ __launch_bounds__(256) void s5_scan_carry(
    const float* __restrict__ cst, const float2* __restrict__ carry,
    float2* __restrict__ prefix)
{
    int p = threadIdx.x, b = blockIdx.x;
    float ar = cst[1536 + p], ai = cst[1792 + p];
    size_t base = (size_t)(b * PDIM + p) * NCHUNK;
    float cr = 0.f, ci = 0.f;
    for (int j = 0; j < NCHUNK; ++j) {
        prefix[base + j] = make_float2(cr, ci);
        float2 v = carry[base + j];
        float nr = ar * cr - ai * ci + v.x;
        float ni = ar * ci + ai * cr + v.y;
        cr = nr; ci = ni;
    }
}

// Scan pass 3: replay chunk with carry-in, emit Xcat[row, p]=re, [row, 256+p]=im
__global__ __launch_bounds__(256) void s5_scan_emit(
    const float* __restrict__ x, const float* __restrict__ cst,
    const float2* __restrict__ prefix, __bf16* __restrict__ Xcat)
{
    int p = threadIdx.x, chunk = blockIdx.x, b = blockIdx.y;
    float lbr = cst[p], lbi = cst[256 + p];
    float vwr = cst[512 + p], vwi = cst[768 + p];
    float vbr = cst[1024 + p], vbi = cst[1280 + p];
    float2 s0 = prefix[(size_t)(b * PDIM + p) * NCHUNK + chunk];
    float sr = s0.x, si = s0.y;
    const float* xp = x + b * SEQ + chunk * CHUNK;
    __bf16* xr = Xcat + (size_t)(b * SEQ + chunk * CHUNK) * 512 + p;
    for (int t = 0; t < CHUNK; ++t) {
        float xv = xp[t];
        float br = fmaf(xv, vwr, vbr);
        float bi = fmaf(xv, vwi, vbi);
        float nr = fmaf(lbr, sr, fmaf(-lbi, si, br));
        float ni = fmaf(lbr, si, fmaf( lbi, sr, bi));
        sr = nr; si = ni;
        xr[0]   = (__bf16)sr;
        xr[256] = (__bf16)si;
        xr += 512;
    }
}

// ---------------------------------------------------------------------------
// WMMA bf16 GEMM: out[M,N] = A[M,K] * B[N,K]^T with fused epilogues.
// A-fragment (16x32): lane<16 row m gets K k..k+7 and k+16..k+23;
//                     lane>=16 gets k+8..k+15 and k+24..k+31 (ISA 7.12.2).
// B-fragment (32x16): lane<16 col n gets K k..k+7 and k+16..k+23; lane>=16 gets k+8..k+15 and k+24..k+31.
// ---------------------------------------------------------------------------
__device__ inline bf16x16 load_a_frag(const __bf16* base, int ld, int row, int k, int lane)
{
    int m = lane & 15, half = lane >> 4;
    const __bf16* p0 = base + (size_t)(row + m) * ld + (k + half * 8);
    union { bf16x16 v; uint4 q[2]; } u;
    u.q[0] = *(const uint4*)p0;
    u.q[1] = *(const uint4*)(p0 + 16);
    return u.v;
}

__device__ inline bf16x16 load_b_frag(const __bf16* base, int ld, int col, int k, int lane)
{
    int n = lane & 15, half = lane >> 4;
    const __bf16* p0 = base + (size_t)(col + n) * ld + (k + half * 8);
    union { bf16x16 v; uint4 q[2]; } u;
    u.q[0] = *(const uint4*)p0;
    u.q[1] = *(const uint4*)(p0 + 16);
    return u.v;
}

// EPI 0: Xff = tanh(acc + D[c]*u) + u,   u = x[row]*l1w[c]+l1b[c]   (bf16 out)
// EPI 1: F   = tanh(acc + bias[c])                                  (bf16 out)
// EPI 2: H3  = acc + bias[c] + res[row,c]                           (f32 out)
template <int EPI>
__global__ __launch_bounds__(256) void gemm_bf16(
    const __bf16* __restrict__ A, const __bf16* __restrict__ B, int K, int ldo,
    void* __restrict__ outp,
    const float* __restrict__ Dv, const float* __restrict__ xin,
    const float* __restrict__ l1w, const float* __restrict__ l1b,
    const float* __restrict__ bias, const __bf16* __restrict__ res)
{
    int lane = threadIdx.x & 31;
    int wv   = threadIdx.x >> 5;                   // 8 waves/block
    int m0 = blockIdx.x * 128 + (wv >> 2) * 64;    // 2 M-waves x 4 N-waves
    int n0 = blockIdx.y * 256 + (wv & 3) * 64;

    int am = lane & 15, ahalf = (lane >> 4) * 8;   // A-frag lane offsets

    floatx8 acc[4][4] = {};
    for (int k = 0; k < K; k += 32) {
        bf16x16 af[4], bfr[4];
#pragma unroll
        for (int i = 0; i < 4; ++i) af[i]  = load_a_frag(A, K, m0 + 16 * i, k, lane);
#pragma unroll
        for (int j = 0; j < 4; ++j) bfr[j] = load_b_frag(B, K, n0 + 16 * j, k, lane);

        // Prefetch next K-step of the streaming A operand (global_prefetch_b8)
        if (k + 32 < K) {
#pragma unroll
            for (int i = 0; i < 4; ++i)
                __builtin_prefetch(A + (size_t)(m0 + 16 * i + am) * K + (k + 32 + ahalf), 0, 1);
        }

#pragma unroll
        for (int i = 0; i < 4; ++i)
#pragma unroll
            for (int j = 0; j < 4; ++j)
                acc[i][j] = __builtin_amdgcn_wmma_f32_16x16x32_bf16(
                    false, af[i], false, bfr[j], (short)0, acc[i][j], false, false);
    }

    // C/D layout: VGPR r -> (lane<16: M=r, N=lane) (lane>=16: M=r+8, N=lane-16)
    int coln  = lane & 15;
    int rowof = (lane >> 4) * 8;
#pragma unroll
    for (int i = 0; i < 4; ++i)
#pragma unroll
        for (int j = 0; j < 4; ++j)
#pragma unroll
            for (int r = 0; r < 8; ++r) {
                int row = m0 + i * 16 + rowof + r;
                int c   = n0 + j * 16 + coln;
                float v = acc[i][j][r];
                if (EPI == 0) {
                    float uu = fmaf(xin[row], l1w[c], l1b[c]);
                    v = fast_tanh(fmaf(Dv[c], uu, v)) + uu;
                    ((__bf16*)outp)[(size_t)row * ldo + c] = (__bf16)v;
                } else if (EPI == 1) {
                    v = fast_tanh(v + bias[c]);
                    ((__bf16*)outp)[(size_t)row * ldo + c] = (__bf16)v;
                } else {
                    v = v + bias[c] + (float)res[(size_t)row * DIMH + c];
                    ((float*)outp)[(size_t)row * ldo + c] = v;
                }
            }
}

// out[row] = sum_h H3[row,h]*l2w[h] + l2b   — one wave32 per row
__global__ __launch_bounds__(256) void l2_reduce(
    const float* __restrict__ H3, const float* __restrict__ w,
    const float* __restrict__ bvec, float* __restrict__ out)
{
    int wv = threadIdx.x >> 5, lane = threadIdx.x & 31;
    int row = blockIdx.x * 8 + wv;
    const float* hp = H3 + (size_t)row * DIMH;
    float s = 0.f;
    for (int h = lane; h < DIMH; h += 32) s = fmaf(hp[h], w[h], s);
    for (int off = 16; off > 0; off >>= 1) s += __shfl_down(s, off, 32);
    if (lane == 0) out[row] = s + bvec[0];
}

// ---------------------------------------------------------------------------
extern "C" void kernel_launch(void* const* d_in, const int* in_sizes, int n_in,
                              void* d_out, int out_size, void* d_ws, size_t ws_size,
                              hipStream_t stream)
{
    const float* x        = (const float*)d_in[0];
    const float* l1_w     = (const float*)d_in[1];
    const float* l1_b     = (const float*)d_in[2];
    const float* lam_re   = (const float*)d_in[3];
    const float* lam_im   = (const float*)d_in[4];
    const float* B_re     = (const float*)d_in[5];
    const float* B_im     = (const float*)d_in[6];
    const float* C_re     = (const float*)d_in[7];
    const float* C_im     = (const float*)d_in[8];
    const float* Dvec     = (const float*)d_in[9];
    const float* log_step = (const float*)d_in[10];
    const float* ff_enc_w = (const float*)d_in[11];
    const float* ff_enc_b = (const float*)d_in[12];
    const float* ff_dec_w = (const float*)d_in[13];
    const float* ff_dec_b = (const float*)d_in[14];
    const float* l2_w     = (const float*)d_in[15];
    const float* l2_b     = (const float*)d_in[16];
    float* out = (float*)d_out;

    // workspace bump allocator
    uintptr_t base = (uintptr_t)d_ws;
    auto alloc = [&](size_t bytes) -> void* {
        void* p = (void*)base;
        base += (bytes + 255) & ~(size_t)255;
        return p;
    };
    float*  cst    = (float*)alloc(8 * 256 * sizeof(float));
    float2* carry  = (float2*)alloc((size_t)BATCH * PDIM * NCHUNK * sizeof(float2));
    float2* prefix = (float2*)alloc((size_t)BATCH * PDIM * NCHUNK * sizeof(float2));
    __bf16* W1     = (__bf16*)alloc((size_t)DIMH * 512 * 2);
    __bf16* Wenc   = (__bf16*)alloc((size_t)HID * DIMH * 2);
    __bf16* Wdec   = (__bf16*)alloc((size_t)DIMH * HID * 2);
    __bf16* Xcat   = (__bf16*)alloc((size_t)ROWS * 512 * 2);
    __bf16* Xff    = (__bf16*)alloc((size_t)ROWS * DIMH * 2);
    __bf16* F      = (__bf16*)alloc((size_t)ROWS * HID * 2);
    float*  H3     = (float*)alloc((size_t)ROWS * DIMH * 4);
    (void)ws_size; (void)n_in; (void)in_sizes; (void)out_size;

    // --- precompute ---
    prep_consts<<<1, 256, 0, stream>>>(lam_re, lam_im, log_step, B_re, B_im,
                                       l1_w, l1_b, cst);
    build_w1<<<(DIMH * 512) / 256, 256, 0, stream>>>(C_re, C_im, W1);
    f32_to_bf16<<<(HID * DIMH) / 256, 256, 0, stream>>>(ff_enc_w, Wenc, HID * DIMH);
    f32_to_bf16<<<(DIMH * HID) / 256, 256, 0, stream>>>(ff_dec_w, Wdec, DIMH * HID);

    // --- chunked constant-coefficient scan ---
    s5_scan_chunks<<<dim3(NCHUNK, BATCH), 256, 0, stream>>>(x, cst, carry);
    s5_scan_carry<<<BATCH, 256, 0, stream>>>(cst, carry, prefix);
    s5_scan_emit<<<dim3(NCHUNK, BATCH), 256, 0, stream>>>(x, cst, prefix, Xcat);

    // --- GEMM 1: C projection + D*u + tanh + residual -> Xff (bf16) ---
    gemm_bf16<0><<<dim3(ROWS / 128, 1), 256, 0, stream>>>(
        Xcat, W1, 512, DIMH, (void*)Xff, Dvec, x, l1_w, l1_b, nullptr, nullptr);

    // --- GEMM 2: ff_enc + tanh -> F (bf16) ---
    gemm_bf16<1><<<dim3(ROWS / 128, HID / 256), 256, 0, stream>>>(
        Xff, Wenc, DIMH, HID, (void*)F, nullptr, nullptr, nullptr, nullptr,
        ff_enc_b, nullptr);

    // --- GEMM 3: ff_dec + bias + residual -> H3 (f32) ---
    gemm_bf16<2><<<dim3(ROWS / 128, 1), 256, 0, stream>>>(
        F, Wdec, HID, DIMH, (void*)H3, nullptr, nullptr, nullptr, nullptr,
        ff_dec_b, Xff);

    // --- l2 projection ---
    l2_reduce<<<ROWS / 8, 256, 0, stream>>>(H3, l2_w, l2_b, out);
}
